// MultiHeadAttention_23158463660795
// MI455X (gfx1250) — matrix operand order repaired and checked by reading, not compiled, not run on hardware
//
#include <hip/hip_runtime.h>

// ---------------------------------------------------------------------------
// MHA + LayerNorm forward for MI455X (gfx1250, wave32, WMMA f16->f32, TDM)
// B=2, S=2048, D_MODEL=1024, H=16, dk=dv=64
// Outputs (concatenated in d_out): out [B,S,1024] fp32, attn [B,H,S,S] fp32
// ---------------------------------------------------------------------------

#define B_   2
#define S_   2048
#define D_   1024
#define H_   16
#define DK_  64
#define M_   (B_ * S_)          // 4096 rows for the big GEMMs
#define NEGV (-1e9f)
#define SCALE 0.125f            // 1/sqrt(64)

typedef __attribute__((ext_vector_type(16))) _Float16 v16h;
typedef __attribute__((ext_vector_type(8)))  float    v8f;
typedef __attribute__((ext_vector_type(4)))  int      v4i;
typedef __attribute__((ext_vector_type(4)))  unsigned int u32x4;
typedef __attribute__((ext_vector_type(8)))  unsigned int u32x8;

union AF { v16h h; v4i i[2]; };

// A fragment: 16x32 f16, row-major source with leading dim `ld` (elements).
// Lane L: row = L&15, half hl = L>>4. Elements 0..7 -> K = 8*hl..8*hl+7,
// elements 8..15 -> K = 16+8*hl .. +7.            (ISA 7.12.2, 16-bit A)
__device__ __forceinline__ v16h load_frag_a(const _Float16* base, int ld, int lane) {
  int row = lane & 15, hl = lane >> 4;
  const _Float16* p = base + (size_t)row * ld + 8 * hl;
  AF u;
  u.i[0] = *(const v4i*)(p);
  u.i[1] = *(const v4i*)(p + 16);
  return u.h;
}

// B fragment: 32x16 f16. Source is Bt[n][k] row-major (B column-major),
// leading dim `ld`. Lane L: col = L&15, elements 0..7 -> K = 8*hl + e,
// elements 8..15 -> K = 16+8*hl + (e-8).
__device__ __forceinline__ v16h load_frag_b(const _Float16* bt, int ld, int lane) {
  int col = lane & 15, hl = lane >> 4;
  const _Float16* p = bt + (size_t)col * ld + 8 * hl;
  AF u;
  u.i[0] = *(const v4i*)(p);
  u.i[1] = *(const v4i*)(p + 16);
  return u.h;
}

__device__ __forceinline__ v8f wmma_f16(v16h a, v16h b, v8f c) {
  return __builtin_amdgcn_wmma_f32_16x16x32_f16(false, a, false, b, (short)0, c,
                                                false, false);
}

// ---------------------------------------------------------------------------
// Tensor Data Mover: one-instruction DMA of a 2D f16 tile (tile_d1 rows x
// tile_d0 elements, row stride stride_d0) from global into LDS at lds_off.
// D# per CDNA5 ISA: group0 {count=1, lds_addr, global_addr,
// type=2}; group1 {data_size=1(2B), tensor dims, tile dims, dim0 stride}.
// 2D tensor -> 2-group form (VADDR2/VADDR3 = NULL). Tracked by TENSORcnt.
__device__ __forceinline__ void tdm_load_tile_2d(unsigned int lds_off,
                                                 const _Float16* gptr,
                                                 unsigned int tensor_d0,
                                                 unsigned int tensor_d1,
                                                 unsigned int stride_d0,
                                                 unsigned int tile_d0,
                                                 unsigned int tile_d1) {
  unsigned long long ga = (unsigned long long)gptr;
  u32x4 g0;
  g0.x = 1u;                                      // count=1, user descriptor
  g0.y = lds_off;                                 // lds_addr [63:32]
  g0.z = (unsigned int)ga;                        // global_addr [95:64]
  g0.w = (unsigned int)(ga >> 32) | (2u << 30);   // addr[56:32] | type=2
  u32x8 g1;
  g1.s0 = (1u << 16);                             // data_size=1 (2 bytes)
  g1.s1 = (tensor_d0 & 0xFFFFu) << 16;            // tensor_dim0 lo16 -> 63:48
  g1.s2 = (tensor_d0 >> 16) | ((tensor_d1 & 0xFFFFu) << 16);  // d0 hi / d1 lo
  g1.s3 = (tensor_d1 >> 16) | (tile_d0 << 16);    // d1 hi | tile_dim0 -> 127:112
  g1.s4 = tile_d1;                                // tile_dim1 (tile_dim2=0)
  g1.s5 = stride_d0;                              // tensor_dim0_stride lo32
  g1.s6 = 0u;
  g1.s7 = 0u;
  asm volatile("tensor_load_to_lds %0, %1" :: "s"(g0), "s"(g1) : "memory");
}

__device__ __forceinline__ unsigned int lds_off_of(const void* p) {
  return (unsigned int)(uintptr_t)p;   // low 32 bits of LDS aperture address
}

// ---------------------------------------------------------------------------
// fp32 -> f16 elementwise
__global__ __launch_bounds__(256) void k_f32_to_f16(const float* __restrict__ in,
                                                    _Float16* __restrict__ out,
                                                    int n) {
  int i = blockIdx.x * 256 + threadIdx.x;
  if (i < n) out[i] = (_Float16)in[i];
}

// W [K][N] fp32 -> Wt [N][K] f16 (so B-fragments read contiguous K)
__global__ __launch_bounds__(256) void k_transpose_f16(const float* __restrict__ in,
                                                       _Float16* __restrict__ out,
                                                       int K, int N) {
  int i = blockIdx.x * 256 + threadIdx.x;
  if (i < K * N) {
    int k = i / N, n = i - k * N;
    out[(size_t)n * K + k] = (_Float16)in[i];
  }
}

// ---------------------------------------------------------------------------
// Projection GEMM: P = X[4096,1024] @ W[1024,1024], f16 out in head layout.
// Block = 4 waves, tile 64x64. K loop in 32-slices, TDM double-buffered LDS.
// mode 0: Out[((b*H+h)*S + s)*64 + d]   (Q, K)
// mode 1: Out[((b*H+h)*64 + d)*S + s]   (V transposed, for PV B-fragments)
__global__ __launch_bounds__(128) void k_gemm_proj(const _Float16* __restrict__ X,
                                                   const _Float16* __restrict__ Wt,
                                                   _Float16* __restrict__ Out,
                                                   int mode) {
  __shared__ _Float16 sA[2][64 * 32];
  __shared__ _Float16 sB[2][64 * 32];
  int lane = threadIdx.x & 31, wave = threadIdx.x >> 5;
  int hl = lane >> 4, cl = lane & 15;
  int mb = blockIdx.y * 64;
  int n0 = blockIdx.x * 64;

  if (wave == 0) {   // one wave drives the TDM; TENSORcnt is per-wave
    tdm_load_tile_2d(lds_off_of(&sA[0][0]), X + (size_t)mb * D_, D_, M_, D_, 32, 64);
    tdm_load_tile_2d(lds_off_of(&sB[0][0]), Wt + (size_t)n0 * D_, D_, D_, D_, 32, 64);
  }

  v8f c[4] = {};
  int buf = 0;
  for (int kk = 0; kk < D_; kk += 32, buf ^= 1) {
    if (wave == 0) {
      if (kk + 32 < D_) {
        tdm_load_tile_2d(lds_off_of(&sA[buf ^ 1][0]),
                         X + (size_t)mb * D_ + kk + 32, D_, M_, D_, 32, 64);
        tdm_load_tile_2d(lds_off_of(&sB[buf ^ 1][0]),
                         Wt + (size_t)n0 * D_ + kk + 32, D_, D_, D_, 32, 64);
        // in-order TENSORcnt: <=2 pending leaves only next-buffer DMAs in flight
        __builtin_amdgcn_s_wait_tensorcnt(2);
      } else {
        __builtin_amdgcn_s_wait_tensorcnt(0);
      }
    }
    __syncthreads();                       // publish buf to all waves
    v16h a = load_frag_a(&sA[buf][wave * 16 * 32], 32, lane);
#pragma unroll
    for (int t = 0; t < 4; ++t) {
      v16h b = load_frag_b(&sB[buf][t * 16 * 32], 32, lane);
      c[t] = wmma_f16(a, b, c[t]);
    }
    __syncthreads();                       // done reading buf before it is reused
  }

#pragma unroll
  for (int t = 0; t < 4; ++t) {
#pragma unroll
    for (int r = 0; r < 8; ++r) {
      int m = mb + wave * 16 + r + 8 * hl;   // global row
      int n = n0 + t * 16 + cl;              // global col
      int b = m >> 11, s = m & (S_ - 1);
      int h = n >> 6, d = n & 63;
      _Float16 v = (_Float16)c[t][r];
      if (mode == 0)
        Out[(((size_t)b * H_ + h) * S_ + s) * DK_ + d] = v;
      else
        Out[(((size_t)b * H_ + h) * DK_ + d) * S_ + s] = v;
    }
  }
}

// ---------------------------------------------------------------------------
// K-fragments for one 32-column score chunk (register double-buffered).
struct KF { v16h b00, b01, b10, b11; };

__device__ __forceinline__ KF load_kf(const _Float16* __restrict__ Kp, int c0,
                                      int lane) {
  KF kf;
  kf.b00 = load_frag_b(Kp + (size_t)(c0)*DK_ + 0,  DK_, lane);
  kf.b01 = load_frag_b(Kp + (size_t)(c0)*DK_ + 32, DK_, lane);
  kf.b10 = load_frag_b(Kp + (size_t)(c0 + 16) * DK_ + 0,  DK_, lane);
  kf.b11 = load_frag_b(Kp + (size_t)(c0 + 16) * DK_ + 32, DK_, lane);
  return kf;
}

__device__ __forceinline__ void score_eval(const KF& kf,
                                           const unsigned char* __restrict__ mp,
                                           int q0, int c0, int lane,
                                           v16h aQ0, v16h aQ1,
                                           v8f& s0, v8f& s1) {
  int hl = lane >> 4, cl = lane & 15;
  v8f z = {};
  s0 = wmma_f16(aQ1, kf.b01, wmma_f16(aQ0, kf.b00, z));
  s1 = wmma_f16(aQ1, kf.b11, wmma_f16(aQ0, kf.b10, z));
#pragma unroll
  for (int r = 0; r < 8; ++r) {
    int q = q0 + r + 8 * hl;
    size_t ro = (size_t)q * S_;
    unsigned char mA = mp[ro + c0 + cl];
    unsigned char mB = mp[ro + c0 + 16 + cl];
    s0[r] = mA ? NEGV : s0[r] * SCALE;
    s1[r] = mB ? NEGV : s1[r] * SCALE;
  }
}

// Attention: grid (S/64, H, B), 4 waves/block, wave owns 16 query rows.
// Pass 1: row max. Pass 2: exp + rowsum + context WMMA (exp staged via LDS to
// convert C-layout -> A-layout). Pass 3: write normalized attn (fp32).
__global__ __launch_bounds__(128) void k_attn(const _Float16* __restrict__ Qh,
                                              const _Float16* __restrict__ Kh,
                                              const _Float16* __restrict__ Vt,
                                              const unsigned char* __restrict__ mask,
                                              float* __restrict__ attn_out,
                                              _Float16* __restrict__ ctx) {
  __shared__ _Float16 eLds[4][16 * 32];   // per-wave 16x32 exp staging tile
  int lane = threadIdx.x & 31, wave = threadIdx.x >> 5;
  int hl = lane >> 4, cl = lane & 15;
  int b = blockIdx.z, h = blockIdx.y;
  int q0 = blockIdx.x * 64 + wave * 16;

  const _Float16* Qp = Qh + ((size_t)b * H_ + h) * S_ * DK_;
  const _Float16* Kp = Kh + ((size_t)b * H_ + h) * S_ * DK_;
  const _Float16* Vp = Vt + ((size_t)b * H_ + h) * DK_ * S_;
  const unsigned char* mp = mask + (size_t)b * S_ * S_;

  v16h aQ0 = load_frag_a(Qp + (size_t)q0 * DK_ + 0,  DK_, lane);
  v16h aQ1 = load_frag_a(Qp + (size_t)q0 * DK_ + 32, DK_, lane);

  // ---- pass 1: row max -------------------------------------------------
  float rm[8];
#pragma unroll
  for (int r = 0; r < 8; ++r) rm[r] = -3.0e38f;
  {
    KF kf = load_kf(Kp, 0, lane);
    for (int c0 = 0; c0 < S_; c0 += 32) {
      KF kfn = kf;
      if (c0 + 32 < S_) kfn = load_kf(Kp, c0 + 32, lane);  // prefetch next
      v8f s0, s1;
      score_eval(kf, mp, q0, c0, lane, aQ0, aQ1, s0, s1);
#pragma unroll
      for (int r = 0; r < 8; ++r) rm[r] = fmaxf(rm[r], fmaxf(s0[r], s1[r]));
      kf = kfn;
    }
  }
#pragma unroll
  for (int r = 0; r < 8; ++r) {
    for (int mxor = 1; mxor < 16; mxor <<= 1)
      rm[r] = fmaxf(rm[r], __shfl_xor(rm[r], mxor));
  }

  // ---- pass 2: exp, rowsum, context accumulation -----------------------
  float rs[8] = {0, 0, 0, 0, 0, 0, 0, 0};
  v8f cc[4] = {};
  {
    KF kf = load_kf(Kp, 0, lane);
    for (int c0 = 0; c0 < S_; c0 += 32) {
      KF kfn = kf;
      if (c0 + 32 < S_) kfn = load_kf(Kp, c0 + 32, lane);
      v8f s0, s1;
      score_eval(kf, mp, q0, c0, lane, aQ0, aQ1, s0, s1);
#pragma unroll
      for (int r = 0; r < 8; ++r) {
        float e0 = __expf(s0[r] - rm[r]);
        float e1 = __expf(s1[r] - rm[r]);
        rs[r] += e0 + e1;
        int row = r + 8 * hl;
        eLds[wave][row * 32 + cl] = (_Float16)e0;
        eLds[wave][row * 32 + 16 + cl] = (_Float16)e1;
      }
      // LDS ops are in-order within a wave (DScnt); tile is wave-private.
      v16h aE = load_frag_a(&eLds[wave][0], 32, lane);
#pragma unroll
      for (int dt = 0; dt < 4; ++dt) {
        v16h bV = load_frag_b(Vp + (size_t)(dt * 16) * S_ + c0, S_, lane);
        cc[dt] = wmma_f16(aE, bV, cc[dt]);
      }
      kf = kfn;
    }
  }
#pragma unroll
  for (int r = 0; r < 8; ++r) {
    for (int mxor = 1; mxor < 16; mxor <<= 1)
      rs[r] += __shfl_xor(rs[r], mxor);
  }

  // context /= rowsum, store f16 as [b, s, h*64+d] for the output GEMM
#pragma unroll
  for (int dt = 0; dt < 4; ++dt) {
#pragma unroll
    for (int r = 0; r < 8; ++r) {
      int q = q0 + r + 8 * hl;
      ctx[((size_t)b * S_ + q) * D_ + h * DK_ + dt * 16 + cl] =
          (_Float16)(cc[dt][r] / rs[r]);
    }
  }

  // ---- pass 3: write normalized attention probabilities ----------------
  float* ap = attn_out + ((size_t)b * H_ + h) * S_ * S_;
  {
    KF kf = load_kf(Kp, 0, lane);
    for (int c0 = 0; c0 < S_; c0 += 32) {
      KF kfn = kf;
      if (c0 + 32 < S_) kfn = load_kf(Kp, c0 + 32, lane);
      v8f s0, s1;
      score_eval(kf, mp, q0, c0, lane, aQ0, aQ1, s0, s1);
#pragma unroll
      for (int r = 0; r < 8; ++r) {
        int q = q0 + r + 8 * hl;
        float inv = 1.0f / rs[r];
        ap[(size_t)q * S_ + c0 + cl]      = __expf(s0[r] - rm[r]) * inv;
        ap[(size_t)q * S_ + c0 + 16 + cl] = __expf(s1[r] - rm[r]) * inv;
      }
      kf = kfn;
    }
  }
}

// ---------------------------------------------------------------------------
// Output projection + residual: Y = ctx @ Wo + input_Q  (fp32 out).
// Same TDM double-buffered structure as k_gemm_proj.
__global__ __launch_bounds__(128) void k_gemm_out(const _Float16* __restrict__ Ctx,
                                                  const _Float16* __restrict__ WoT,
                                                  const float* __restrict__ resid,
                                                  float* __restrict__ Y) {
  __shared__ _Float16 sA[2][64 * 32];
  __shared__ _Float16 sB[2][64 * 32];
  int lane = threadIdx.x & 31, wave = threadIdx.x >> 5;
  int hl = lane >> 4, cl = lane & 15;
  int mb = blockIdx.y * 64;
  int n0 = blockIdx.x * 64;

  if (wave == 0) {
    tdm_load_tile_2d(lds_off_of(&sA[0][0]), Ctx + (size_t)mb * D_, D_, M_, D_, 32, 64);
    tdm_load_tile_2d(lds_off_of(&sB[0][0]), WoT + (size_t)n0 * D_, D_, D_, D_, 32, 64);
  }

  v8f c[4] = {};
  int buf = 0;
  for (int kk = 0; kk < D_; kk += 32, buf ^= 1) {
    if (wave == 0) {
      if (kk + 32 < D_) {
        tdm_load_tile_2d(lds_off_of(&sA[buf ^ 1][0]),
                         Ctx + (size_t)mb * D_ + kk + 32, D_, M_, D_, 32, 64);
        tdm_load_tile_2d(lds_off_of(&sB[buf ^ 1][0]),
                         WoT + (size_t)n0 * D_ + kk + 32, D_, D_, D_, 32, 64);
        __builtin_amdgcn_s_wait_tensorcnt(2);
      } else {
        __builtin_amdgcn_s_wait_tensorcnt(0);
      }
    }
    __syncthreads();
    v16h a = load_frag_a(&sA[buf][wave * 16 * 32], 32, lane);
#pragma unroll
    for (int t = 0; t < 4; ++t) {
      v16h b = load_frag_b(&sB[buf][t * 16 * 32], 32, lane);
      c[t] = wmma_f16(a, b, c[t]);
    }
    __syncthreads();
  }

#pragma unroll
  for (int t = 0; t < 4; ++t) {
#pragma unroll
    for (int r = 0; r < 8; ++r) {
      size_t m = mb + wave * 16 + r + 8 * hl;
      size_t n = n0 + t * 16 + cl;
      Y[m * D_ + n] = c[t][r] + resid[m * D_ + n];
    }
  }
}

// Row LayerNorm (gamma=1, beta=0, eps=1e-5): one block per row of 1024.
__global__ __launch_bounds__(256) void k_ln(const float* __restrict__ Y,
                                            float* __restrict__ Out) {
  __shared__ float red[8];
  __shared__ float stat;
  int row = blockIdx.x;
  const float* y = Y + (size_t)row * D_;
  int lane = threadIdx.x & 31, wv = threadIdx.x >> 5;
  float v0 = y[threadIdx.x], v1 = y[threadIdx.x + 256];
  float v2 = y[threadIdx.x + 512], v3 = y[threadIdx.x + 768];
  float s = v0 + v1 + v2 + v3;
  for (int mxor = 1; mxor < 32; mxor <<= 1) s += __shfl_xor(s, mxor);
  if (!lane) red[wv] = s;
  __syncthreads();
  if (threadIdx.x == 0) {
    float t = 0;
    for (int i = 0; i < 8; ++i) t += red[i];
    stat = t * (1.0f / D_);
  }
  __syncthreads();
  float mu = stat;
  float d0 = v0 - mu, d1 = v1 - mu, d2 = v2 - mu, d3 = v3 - mu;
  float vs = d0 * d0 + d1 * d1 + d2 * d2 + d3 * d3;
  for (int mxor = 1; mxor < 32; mxor <<= 1) vs += __shfl_xor(vs, mxor);
  __syncthreads();
  if (!lane) red[wv] = vs;
  __syncthreads();
  if (threadIdx.x == 0) {
    float t = 0;
    for (int i = 0; i < 8; ++i) t += red[i];
    stat = rsqrtf(t * (1.0f / D_) + 1e-5f);
  }
  __syncthreads();
  float r = stat;
  float* o = Out + (size_t)row * D_;
  o[threadIdx.x] = d0 * r;
  o[threadIdx.x + 256] = d1 * r;
  o[threadIdx.x + 512] = d2 * r;
  o[threadIdx.x + 768] = d3 * r;
}

// ---------------------------------------------------------------------------
extern "C" void kernel_launch(void* const* d_in, const int* in_sizes, int n_in,
                              void* d_out, int out_size, void* d_ws, size_t ws_size,
                              hipStream_t stream) {
  (void)in_sizes; (void)n_in; (void)out_size; (void)ws_size;
  const float* inQ = (const float*)d_in[0];
  const float* inK = (const float*)d_in[1];
  const float* inV = (const float*)d_in[2];
  // jax bool array -> 1 byte per element
  const unsigned char* mask = (const unsigned char*)d_in[3];
  const float* WQ = (const float*)d_in[4];
  const float* WK = (const float*)d_in[5];
  const float* WV = (const float*)d_in[6];
  const float* WO = (const float*)d_in[7];

  float* out  = (float*)d_out;                       // [B,S,1024]
  float* attn = out + (size_t)B_ * S_ * D_;          // [B,H,S,S]

  char* ws = (char*)d_ws;
  const size_t MB = 1u << 20;
  _Float16* xq  = (_Float16*)(ws + 0 * MB);          // 8 MiB each
  _Float16* xk  = (_Float16*)(ws + 8 * MB);
  _Float16* xv  = (_Float16*)(ws + 16 * MB);
  _Float16* wqT = (_Float16*)(ws + 24 * MB);         // 2 MiB each
  _Float16* wkT = (_Float16*)(ws + 26 * MB);
  _Float16* wvT = (_Float16*)(ws + 28 * MB);
  _Float16* woT = (_Float16*)(ws + 30 * MB);
  _Float16* qh  = (_Float16*)(ws + 32 * MB);         // [B,H,S,64]
  _Float16* kh  = (_Float16*)(ws + 40 * MB);         // [B,H,S,64]
  _Float16* vt  = (_Float16*)(ws + 48 * MB);         // [B,H,64,S]
  _Float16* ctx = (_Float16*)(ws + 56 * MB);         // [B,S,1024]
  float*    yb  = (float*)   (ws + 64 * MB);         // [B*S,1024] fp32

  const int nX = B_ * S_ * D_;                       // 4,194,304
  const int nW = D_ * D_;                            // 1,048,576

  k_f32_to_f16<<<(nX + 255) / 256, 256, 0, stream>>>(inQ, xq, nX);
  k_f32_to_f16<<<(nX + 255) / 256, 256, 0, stream>>>(inK, xk, nX);
  k_f32_to_f16<<<(nX + 255) / 256, 256, 0, stream>>>(inV, xv, nX);
  k_transpose_f16<<<(nW + 255) / 256, 256, 0, stream>>>(WQ, wqT, D_, D_);
  k_transpose_f16<<<(nW + 255) / 256, 256, 0, stream>>>(WK, wkT, D_, D_);
  k_transpose_f16<<<(nW + 255) / 256, 256, 0, stream>>>(WV, wvT, D_, D_);
  k_transpose_f16<<<(nW + 255) / 256, 256, 0, stream>>>(WO, woT, D_, D_);

  dim3 gGemm(D_ / 64, M_ / 64);                      // (16, 64)
  k_gemm_proj<<<gGemm, 128, 0, stream>>>(xq, wqT, qh, 0);
  k_gemm_proj<<<gGemm, 128, 0, stream>>>(xk, wkT, kh, 0);
  k_gemm_proj<<<gGemm, 128, 0, stream>>>(xv, wvT, vt, 1);

  dim3 gAttn(S_ / 64, H_, B_);                       // (32, 16, 2)
  k_attn<<<gAttn, 128, 0, stream>>>(qh, kh, vt, mask, attn, ctx);

  k_gemm_out<<<gGemm, 128, 0, stream>>>(ctx, woT, inQ, yb);
  k_ln<<<M_, 256, 0, stream>>>(yb, out);
}
